// SparseDistributed_89807766159381
// MI455X (gfx1250) — compile-verified, matrix operand order checked by reading, not run on hardware
//
#include <hip/hip_runtime.h>

#define NUM_ADDR    100000
#define ADDR_DIM    1024
#define CONTENT_DIM 1024
#define BATCH       512
#define KCHUNK      128          // fp8 WMMA K depth
#define NTILE       64           // N columns per workgroup
#define APITCH_DW   36           // LDS row pitch in dwords (128B data + 16B pad, keeps 16B align)

typedef __attribute__((ext_vector_type(16))) int       v16i;
typedef __attribute__((ext_vector_type(8)))  float     v8f;
typedef __attribute__((ext_vector_type(4)))  unsigned  uvec4;

// +-1.0f -> FP8 E4M3 (+1 = 0x38, -1 = 0xB8): keep the sign bit, force exponent=bias.
__device__ __forceinline__ unsigned pack_fp8x4(float4 f) {
  unsigned s0 = (__float_as_uint(f.x) >> 24) & 0x80u;
  unsigned s1 = (__float_as_uint(f.y) >> 24) & 0x80u;
  unsigned s2 = (__float_as_uint(f.z) >> 24) & 0x80u;
  unsigned s3 = (__float_as_uint(f.w) >> 24) & 0x80u;
  return 0x38383838u | s0 | (s1 << 8) | (s2 << 16) | (s3 << 24);
}

// Destination dword index (within a 32-dword 128-K block) for the float4 covering
// K = q*4 .. q*4+3, permuted so each half-lane's 16 fragment dwords are contiguous.
__device__ __forceinline__ int perm_dw(int q) {
  return ((q >> 1) & 1) * 16 + (q >> 2) * 2 + (q & 1);
}

__device__ __forceinline__ v16i make_frag16(uvec4 a, uvec4 b, uvec4 c, uvec4 d) {
  union { uvec4 u[4]; v16i v; } t;
  t.u[0] = a; t.u[1] = b; t.u[2] = c; t.u[3] = d;
  return t.v;
}

__global__ __launch_bounds__(256)
void sdm_init(float* __restrict__ accum, int* __restrict__ cnt) {
  int i = blockIdx.x * 256 + threadIdx.x;
  if (i < BATCH * CONTENT_DIM) accum[i] = 0.0f;
  if (i == 0) *cnt = 0;
}

// Convert the [512 x 1024] +-1 query matrix to FP8, rows stored in the
// fragment-ready permuted order per 128-K block. One thread -> one float4 -> one dword.
__global__ __launch_bounds__(256)
void sdm_cvt_query(const float* __restrict__ q, unsigned* __restrict__ qf8) {
  int i = blockIdx.x * 256 + threadIdx.x;        // dword id, 256 dwords per row
  if (i < BATCH * ADDR_DIM / 4) {
    int row = i >> 8;
    int d   = i & 255;
    int blk = d >> 5;                            // which 128-K block
    int qq  = d & 31;                            // float4 within block
    float4 f = ((const float4*)q)[i];
    qf8[row * (ADDR_DIM / 4) + blk * 32 + perm_dw(qq)] = pack_fp8x4(f);
  }
}

__global__ __launch_bounds__(256)
void sdm_sim(const unsigned* __restrict__ qf8, const float* __restrict__ addrs,
             const int* __restrict__ thr_p, int* __restrict__ pair_cnt,
             int2* __restrict__ pairs, int pair_cap)
{
  __shared__ __attribute__((aligned(16))) unsigned shA[NTILE * APITCH_DW];

  const int tid   = threadIdx.x;
  const int wave  = tid >> 5;          // 8 waves of 32 (wave32)
  const int lane  = tid & 31;
  const int hsel  = lane >> 4;         // half-lane K interleave select
  const int l16   = lane & 15;
  const int nbase = blockIdx.x * NTILE;
  const float thr = (float)(*thr_p);

  v8f acc[16] = {};                    // acc[nb*4 + mb]: 64(M) x 64(N) per wave
  const int mwave = wave * 64;

  for (int k0 = 0; k0 < ADDR_DIM; k0 += KCHUNK) {
    __syncthreads();
    // Stage the 64 x 128 addresses tile into LDS as permuted FP8 (single HBM pass).
    // 64 rows x 32 float4 = 2048 float4 loads, 8 per thread.
    #pragma unroll
    for (int it = 0; it < 8; ++it) {
      int j   = tid + it * 256;
      int row = j >> 5;
      int q   = j & 31;
      int ga  = nbase + row;
      float4 f = make_float4(0.f, 0.f, 0.f, 0.f);   // pad rows pack to FP8 +1; dropped by n < NUM_ADDR
      if (ga < NUM_ADDR)
        f = *(const float4*)(addrs + (size_t)ga * ADDR_DIM + k0 + q * 4);
      shA[row * APITCH_DW + perm_dw(q)] = pack_fp8x4(f);
    }
    if (k0 + KCHUNK < ADDR_DIM) {
      int row = tid >> 2;                           // rows 0..63, 4 cacheline hints each
      int ga  = nbase + row;
      if (ga < NUM_ADDR)
        __builtin_prefetch(addrs + (size_t)ga * ADDR_DIM + k0 + KCHUNK + (tid & 3) * 32, 0, 0);
    }
    __syncthreads();

    // A fragments: 4 blocks of 16 query rows, 64 contiguous bytes per lane from the
    // L2-resident permuted FP8 query buffer (lanes 0-15: first half, 16-31: second).
    v16i afrag[4];
    #pragma unroll
    for (int mb = 0; mb < 4; ++mb) {
      int m = mwave + mb * 16 + l16;
      const uvec4* ap = (const uvec4*)(qf8 + (size_t)m * (ADDR_DIM / 4) + (k0 >> 2) + hsel * 16);
      afrag[mb] = make_frag16(ap[0], ap[1], ap[2], ap[3]);
    }
    #pragma unroll
    for (int nb = 0; nb < 4; ++nb) {
      const uvec4* bp = (const uvec4*)&shA[(nb * 16 + l16) * APITCH_DW + hsel * 16];
      v16i bfrag = make_frag16(bp[0], bp[1], bp[2], bp[3]);
      #pragma unroll
      for (int mb = 0; mb < 4; ++mb) {
        acc[nb * 4 + mb] = __builtin_amdgcn_wmma_f32_16x16x128_fp8_fp8(
            afrag[mb], bfrag, (short)0, acc[nb * 4 + mb], false, false);
      }
    }
  }

  // Threshold + emit active (b, n) pairs. C layout: VGPR r -> M = r + 8*hsel, N = lane&15.
  #pragma unroll
  for (int nb = 0; nb < 4; ++nb) {
    int n = nbase + nb * 16 + l16;
    #pragma unroll
    for (int mb = 0; mb < 4; ++mb) {
      #pragma unroll
      for (int r = 0; r < 8; ++r) {
        float v = acc[nb * 4 + mb][r];
        if (v >= thr && n < NUM_ADDR) {
          int b = mwave + mb * 16 + r + hsel * 8;
          int idx = atomicAdd(pair_cnt, 1);
          if (idx < pair_cap) pairs[idx] = make_int2(b, n);
        }
      }
    }
  }
}

__global__ __launch_bounds__(256)
void sdm_scatter(const float* __restrict__ content, const int* __restrict__ pair_cnt,
                 const int2* __restrict__ pairs, int pair_cap, float* __restrict__ accum)
{
  int np = *pair_cnt;
  if (np > pair_cap) np = pair_cap;
  const int c = threadIdx.x * 4;       // 256 threads x float4 = 1024 cols
  for (int p = blockIdx.x; p < np; p += gridDim.x) {
    int2 pr = pairs[p];
    float4 v = *(const float4*)(content + (size_t)pr.y * CONTENT_DIM + c);
    float* dst = accum + (size_t)pr.x * CONTENT_DIM + c;
    atomicAdd(dst + 0, v.x);
    atomicAdd(dst + 1, v.y);
    atomicAdd(dst + 2, v.z);
    atomicAdd(dst + 3, v.w);
  }
}

__global__ __launch_bounds__(256)
void sdm_sign(const float* __restrict__ accum, float* __restrict__ out) {
  int i = blockIdx.x * 256 + threadIdx.x;
  if (i < BATCH * CONTENT_DIM) {
    float x = accum[i];
    out[i] = (x > 0.f) ? 1.f : ((x < 0.f) ? -1.f : 0.f);
  }
}

extern "C" void kernel_launch(void* const* d_in, const int* in_sizes, int n_in,
                              void* d_out, int out_size, void* d_ws, size_t ws_size,
                              hipStream_t stream)
{
  (void)in_sizes; (void)n_in; (void)out_size;
  const float* query   = (const float*)d_in[0];   // [512,1024] f32 (+-1)
  const float* addrs   = (const float*)d_in[1];   // [100000,1024] f32 (+-1)
  const float* content = (const float*)d_in[2];   // [100000,1024] f32
  const int*   thr     = (const int*)d_in[3];     // scalar int 110
  float* out = (float*)d_out;                     // [512,1024] f32

  char* ws = (char*)d_ws;
  float*    accum = (float*)ws;                                         // 2 MB f32 accumulators
  unsigned* qf8   = (unsigned*)(ws + (size_t)BATCH * CONTENT_DIM * 4);  // 0.5 MB fp8 queries
  int*      cnt   = (int*)(ws + 2ull * 1024 * 1024 + 512 * 1024);       // pair counter
  int2*     pairs = (int2*)(ws + 2ull * 1024 * 1024 + 512 * 1024 + 16); // pair list

  long cap_l = ((long)ws_size - (2L * 1024 * 1024 + 512 * 1024 + 16)) / (long)sizeof(int2);
  int pair_cap = cap_l < 0 ? 0 : (cap_l > (1 << 22) ? (1 << 22) : (int)cap_l);

  sdm_init<<<dim3((BATCH * CONTENT_DIM + 255) / 256), dim3(256), 0, stream>>>(accum, cnt);
  sdm_cvt_query<<<dim3((BATCH * ADDR_DIM / 4 + 255) / 256), dim3(256), 0, stream>>>(query, qf8);
  sdm_sim<<<dim3((NUM_ADDR + NTILE - 1) / NTILE), dim3(256), 0, stream>>>(
      qf8, addrs, thr, cnt, pairs, pair_cap);
  sdm_scatter<<<dim3(2048), dim3(256), 0, stream>>>(content, cnt, pairs, pair_cap, accum);
  sdm_sign<<<dim3((BATCH * CONTENT_DIM + 255) / 256), dim3(256), 0, stream>>>(accum, out);
}
